// MoELayer_33432025432260
// MI455X (gfx1250) — matrix operand order repaired and checked by reading, not compiled, not run on hardware
//
#include <hip/hip_runtime.h>
#include <hip/hip_bf16.h>
#include <math.h>

typedef __attribute__((ext_vector_type(16))) __bf16 v16bf;
typedef __attribute__((ext_vector_type(8)))  float  v8f;

#define Bb 8
#define Ll 2048
#define Dd 1024
#define Ee 8
#define Hh 512
#define Nn (Bb*Ll)          /* 16384 tokens */
#define LN_EPS 1e-5f
#define XS 1032             /* padded LDS row stride (ushorts) for x tile */
#define HS 520              /* padded LDS row stride (ushorts) for h tile */
#define MT 64               /* token rows per expert block */
#define TPB 512             /* 16 waves */

__device__ __forceinline__ unsigned short f2bf(float f) {
    unsigned u = __float_as_uint(f);
    u += 0x7FFFu + ((u >> 16) & 1u);   // round-to-nearest-even
    return (unsigned short)(u >> 16);
}

// ---------------- x -> bf16 ----------------
__global__ void __launch_bounds__(256) cvt_x_kernel(const float* __restrict__ x,
                                                    unsigned short* __restrict__ xb) {
    size_t i = ((size_t)blockIdx.x * 256 + threadIdx.x) * 4;
    float4 v = *(const float4*)(x + i);
    ushort4 o;
    o.x = f2bf(v.x); o.y = f2bf(v.y); o.z = f2bf(v.z); o.w = f2bf(v.w);
    *(ushort4*)(xb + i) = o;
}

// ------- pack weights into WMMA B-fragment order (bf16) -------
// layout: wp[(((e*NT + nt)*KC + kc) << 9) + lane*16 + j]
//   lane: k-half = lane>>4, n = lane&15 ;  element j -> k_in = (lane>>4)*16 + j
__global__ void __launch_bounds__(256) pack_w_kernel(const float* __restrict__ w,
                                                     unsigned short* __restrict__ wp,
                                                     int KC, int NT, int Kdim, int Ndim) {
    long idx = (long)blockIdx.x * 256 + threadIdx.x;
    int j    = (int)(idx & 15);
    int lane = (int)((idx >> 4) & 31);
    long rest = idx >> 9;
    int kc = (int)(rest % KC); rest /= KC;
    int nt = (int)(rest % NT);
    int e  = (int)(rest / NT);
    int kin = (lane >> 4) * 16 + j;
    int nin = lane & 15;
    int kg = kc * 32 + kin;
    int ng = nt * 16 + nin;
    wp[idx] = f2bf(w[((long)e * Kdim + kg) * Ndim + ng]);
}

// ---------------- gate: LN -> logits -> softmax -> top2 routing ----------------
__global__ void __launch_bounds__(256) gate_kernel(
    const float* __restrict__ x, const float* __restrict__ ln_w, const float* __restrict__ ln_b,
    const float* __restrict__ gate_w, const float* __restrict__ gate_b,
    float* __restrict__ gate_out, int* __restrict__ cnt,
    int* __restrict__ tok_list, float* __restrict__ coef_list)
{
    __shared__ float s_red[8];
    __shared__ float s_bcast[2];
    __shared__ float s_log[8][8];

    int t = threadIdx.x, n = blockIdx.x;
    int wave = t >> 5, lane = t & 31;
    float4 xv = ((const float4*)(x + (size_t)n * Dd))[t];

    // pass 1: mean
    float s = xv.x + xv.y + xv.z + xv.w;
    for (int o = 16; o >= 1; o >>= 1) s += __shfl_xor(s, o, 32);
    if (lane == 0) s_red[wave] = s;
    __syncthreads();
    if (t == 0) { float a = 0.f; for (int w = 0; w < 8; ++w) a += s_red[w]; s_bcast[0] = a * (1.0f / Dd); }
    __syncthreads();
    float mu = s_bcast[0];

    // pass 2: variance
    float d0 = xv.x - mu, d1 = xv.y - mu, d2 = xv.z - mu, d3 = xv.w - mu;
    float q = d0 * d0 + d1 * d1 + d2 * d2 + d3 * d3;
    for (int o = 16; o >= 1; o >>= 1) q += __shfl_xor(q, o, 32);
    if (lane == 0) s_red[wave] = q;
    __syncthreads();
    if (t == 0) { float a = 0.f; for (int w = 0; w < 8; ++w) a += s_red[w]; s_bcast[1] = a * (1.0f / Dd); }
    __syncthreads();
    float rs = rsqrtf(s_bcast[1] + LN_EPS);

    // logits partials (each thread owns elements 4t..4t+3)
    float p[8];
    #pragma unroll
    for (int e2 = 0; e2 < 8; ++e2) p[e2] = 0.f;
    float xs4[4] = { xv.x, xv.y, xv.z, xv.w };
    #pragma unroll
    for (int c = 0; c < 4; ++c) {
        int i = 4 * t + c;
        float xn = (xs4[c] - mu) * rs * ln_w[i] + ln_b[i];
        const float4* gw = (const float4*)(gate_w + (size_t)i * Ee);
        float4 g0 = gw[0], g1 = gw[1];
        p[0] += xn * g0.x; p[1] += xn * g0.y; p[2] += xn * g0.z; p[3] += xn * g0.w;
        p[4] += xn * g1.x; p[5] += xn * g1.y; p[6] += xn * g1.z; p[7] += xn * g1.w;
    }
    #pragma unroll
    for (int e2 = 0; e2 < 8; ++e2)
        for (int o = 16; o >= 1; o >>= 1) p[e2] += __shfl_xor(p[e2], o, 32);
    if (lane == 0) { for (int e2 = 0; e2 < 8; ++e2) s_log[wave][e2] = p[e2]; }
    __syncthreads();

    if (t == 0) {
        float lg[8];
        for (int e2 = 0; e2 < 8; ++e2) {
            float a = gate_b[e2];
            for (int w = 0; w < 8; ++w) a += s_log[w][e2];
            lg[e2] = a;
        }
        float mx = lg[0];
        for (int e2 = 1; e2 < 8; ++e2) mx = fmaxf(mx, lg[e2]);
        float ex[8], ssum = 0.f;
        for (int e2 = 0; e2 < 8; ++e2) { ex[e2] = expf(lg[e2] - mx); ssum += ex[e2]; }
        float inv = 1.0f / ssum;
        for (int e2 = 0; e2 < 8; ++e2) { ex[e2] *= inv; gate_out[(size_t)n * Ee + e2] = ex[e2]; }
        // top-2 (first occurrence wins ties, matches lax.top_k)
        int i0 = 0;
        for (int e2 = 1; e2 < 8; ++e2) if (ex[e2] > ex[i0]) i0 = e2;
        int i1 = (i0 == 0) ? 1 : 0;
        for (int e2 = 0; e2 < 8; ++e2) if (e2 != i0 && ex[e2] > ex[i1]) i1 = e2;
        float wsum = ex[i0] + ex[i1];
        float c0 = ex[i0] / wsum, c1 = ex[i1] / wsum;
        int s0 = atomicAdd(&cnt[i0], 1);
        tok_list[i0 * Nn + s0] = n; coef_list[i0 * Nn + s0] = c0;
        int s1 = atomicAdd(&cnt[i1], 1);
        tok_list[i1 * Nn + s1] = n; coef_list[i1 * Nn + s1] = c1;
    }
}

// ---------------- fused expert FFN: gather -> GEMM -> GELU -> GEMM -> scatter ----------------
// MT=64 rows/block, 16 waves; each loaded B fragment is reused across 4 M-subtiles.
__global__ void __launch_bounds__(TPB) expert_kernel(
    const unsigned short* __restrict__ xb,
    const unsigned short* __restrict__ w1p,
    const unsigned short* __restrict__ w2p,
    const float* __restrict__ b1, const float* __restrict__ b2,
    const int* __restrict__ cnt, const int* __restrict__ tok_list,
    const float* __restrict__ coef_list, float* __restrict__ out)
{
    __shared__ unsigned short s_x[MT * XS];
    __shared__ unsigned short s_h[MT * HS];
    __shared__ int   s_tok[MT];
    __shared__ float s_coef[MT];

    int e = blockIdx.x;
    int base = blockIdx.y * MT;
    int count = cnt[e];
    if (base >= count) return;
    int valid = min(MT, count - base);
    int t = threadIdx.x;

    if (t < MT) {
        int tk = 0; float cf = 0.f;
        if (t < valid) { tk = tok_list[e * Nn + base + t]; cf = coef_list[e * Nn + base + t]; }
        s_tok[t] = tk; s_coef[t] = cf;
    }
    __syncthreads();

    // gather MT token rows (bf16, 128 uint4 per row), padded LDS stride
    for (int idx = t; idx < MT * 128; idx += TPB) {
        int row = idx >> 7, qq = idx & 127;
        const uint4* src = (const uint4*)(xb + (size_t)s_tok[row] * Dd) + qq;
        *(uint4*)&s_x[row * XS + qq * 8] = *src;
    }
    __syncthreads();

    int wave = t >> 5, lane = t & 31;          // 16 waves
    int hv = lane >> 4, m = lane & 15;         // k-half and row/col index
    union AB { v16bf v; unsigned int w[8]; };

    // ---- Stage A: h[MT,512] = gelu(x @ w1[e] + b1[e]) ----
    for (int nt = wave; nt < (Hh / 16); nt += 16) {
        v8f acc[4];
        #pragma unroll
        for (int su = 0; su < 4; ++su)
            acc[su] = (v8f){ 0.f, 0.f, 0.f, 0.f, 0.f, 0.f, 0.f, 0.f };
        #pragma unroll 4
        for (int kc = 0; kc < (Dd / 32); ++kc) {
            AB b;
            const uint4* bp = (const uint4*)(w1p + ((((size_t)e * 32 + nt) * 32 + kc) << 9) + lane * 16);
            *(uint4*)&b.w[0] = bp[0];
            *(uint4*)&b.w[4] = bp[1];
            #pragma unroll
            for (int su = 0; su < 4; ++su) {
                AB a;
                const unsigned short* xr = &s_x[(su * 16 + m) * XS + kc * 32 + hv * 16];
                #pragma unroll
                for (int p2 = 0; p2 < 4; ++p2) a.w[p2]     = *(const unsigned int*)(xr + 2 * p2);
                #pragma unroll
                for (int p2 = 0; p2 < 4; ++p2) a.w[4 + p2] = *(const unsigned int*)(xr + 8 + 2 * p2);
                acc[su] = __builtin_amdgcn_wmma_f32_16x16x32_bf16(false, a.v, false, b.v,
                                                                  (short)0, acc[su], false, false);
            }
        }
        #pragma unroll
        for (int su = 0; su < 4; ++su) {
            #pragma unroll
            for (int r = 0; r < 8; ++r) {
                int row = su * 16 + hv * 8 + r;
                int col = nt * 16 + m;
                float v = acc[su][r] + b1[e * Hh + col];
                v = 0.5f * v * (1.0f + erff(v * 0.70710678118654752f));   // exact GELU
                s_h[row * HS + col] = f2bf(v);
            }
        }
    }
    __syncthreads();

    // ---- Stage B: y[MT,1024] = h @ w2[e] + b2[e]; scale & scatter ----
    for (int nt = wave; nt < (Dd / 16); nt += 16) {
        v8f acc[4];
        #pragma unroll
        for (int su = 0; su < 4; ++su)
            acc[su] = (v8f){ 0.f, 0.f, 0.f, 0.f, 0.f, 0.f, 0.f, 0.f };
        #pragma unroll 4
        for (int kc = 0; kc < (Hh / 32); ++kc) {
            AB b;
            const uint4* bp = (const uint4*)(w2p + ((((size_t)e * 64 + nt) * 16 + kc) << 9) + lane * 16);
            *(uint4*)&b.w[0] = bp[0];
            *(uint4*)&b.w[4] = bp[1];
            #pragma unroll
            for (int su = 0; su < 4; ++su) {
                AB a;
                const unsigned short* hr = &s_h[(su * 16 + m) * HS + kc * 32 + hv * 16];
                #pragma unroll
                for (int p2 = 0; p2 < 4; ++p2) a.w[p2]     = *(const unsigned int*)(hr + 2 * p2);
                #pragma unroll
                for (int p2 = 0; p2 < 4; ++p2) a.w[4 + p2] = *(const unsigned int*)(hr + 8 + 2 * p2);
                acc[su] = __builtin_amdgcn_wmma_f32_16x16x32_bf16(false, a.v, false, b.v,
                                                                  (short)0, acc[su], false, false);
            }
        }
        #pragma unroll
        for (int su = 0; su < 4; ++su) {
            #pragma unroll
            for (int r = 0; r < 8; ++r) {
                int row = su * 16 + hv * 8 + r;
                if (row < valid) {
                    int col = nt * 16 + m;
                    float v = (acc[su][r] + b2[e * Dd + col]) * s_coef[row];
                    atomicAdd(&out[(size_t)s_tok[row] * Dd + col], v);
                }
            }
        }
    }
}

extern "C" void kernel_launch(void* const* d_in, const int* in_sizes, int n_in,
                              void* d_out, int out_size, void* d_ws, size_t ws_size,
                              hipStream_t stream) {
    const float* x      = (const float*)d_in[0];
    const float* ln_w   = (const float*)d_in[1];
    const float* ln_b   = (const float*)d_in[2];
    const float* gate_w = (const float*)d_in[3];
    const float* gate_b = (const float*)d_in[4];
    const float* w1     = (const float*)d_in[5];
    const float* b1     = (const float*)d_in[6];
    const float* w2     = (const float*)d_in[7];
    const float* b2     = (const float*)d_in[8];

    float* out      = (float*)d_out;                 // [N, D]
    float* gate_out = out + (size_t)Nn * Dd;         // [N, E]

    char* ws = (char*)d_ws;
    size_t off = 0;
    auto take = [&](size_t bytes) { size_t p = off; off = (off + bytes + 255) & ~(size_t)255; return p; };
    unsigned short* xb   = (unsigned short*)(ws + take((size_t)Nn * Dd * 2));
    unsigned short* w1p  = (unsigned short*)(ws + take((size_t)Ee * Dd * Hh * 2));
    unsigned short* w2p  = (unsigned short*)(ws + take((size_t)Ee * Hh * Dd * 2));
    int*            cntp = (int*)           (ws + take((size_t)Ee * sizeof(int)));
    int*            tokl = (int*)           (ws + take((size_t)Ee * Nn * sizeof(int)));
    float*          cofl = (float*)         (ws + take((size_t)Ee * Nn * sizeof(float)));

    hipMemsetAsync(out, 0, (size_t)Nn * Dd * sizeof(float), stream);   // atomics accumulate into it
    hipMemsetAsync(cntp, 0, Ee * sizeof(int), stream);

    cvt_x_kernel <<<(Nn * (Dd / 4)) / 256, 256, 0, stream>>>(x, xb);
    pack_w_kernel<<<(Ee * Dd * Hh) / 256, 256, 0, stream>>>(w1, w1p, /*KC*/32, /*NT*/32, Dd, Hh);
    pack_w_kernel<<<(Ee * Hh * Dd) / 256, 256, 0, stream>>>(w2, w2p, /*KC*/16, /*NT*/64, Hh, Dd);
    gate_kernel  <<<Nn, 256, 0, stream>>>(x, ln_w, ln_b, gate_w, gate_b,
                                          gate_out, cntp, tokl, cofl);
    expert_kernel<<<dim3(Ee, Nn / MT), TPB, 0, stream>>>(xb, w1p, w2p, b1, b2,
                                                         cntp, tokl, cofl, out);
}
